// DeepSeekMoE_76476187673233
// MI455X (gfx1250) — matrix operand order repaired and checked by reading, not compiled, not run on hardware
//
#include <hip/hip_runtime.h>
#include <hip/hip_bf16.h>
#include <stdint.h>

// ---------------------------------------------------------------------------
// Types / WMMA helpers (CDNA5 wave32, v_wmma_f32_16x16x32_bf16)
// ---------------------------------------------------------------------------
typedef __bf16 bf16_t;
typedef __attribute__((ext_vector_type(16))) __bf16 bf16x16;
typedef __attribute__((ext_vector_type(8)))  float  f32x8;
typedef unsigned int u32x4 __attribute__((ext_vector_type(4)));
typedef int          i32x8 __attribute__((ext_vector_type(8)));
typedef int          i32x4 __attribute__((ext_vector_type(4)));

#define T_TOK   4096
#define DMODEL  1024
#define FFDIM   2048
#define NEXP    8

__device__ __forceinline__ f32x8 wmma_bf16(bf16x16 a, bf16x16 b, f32x8 c) {
  return __builtin_amdgcn_wmma_f32_16x16x32_bf16(
      /*neg_a=*/false, a, /*neg_b=*/false, b,
      /*c_mod=*/(short)0, c, /*reuse_a=*/false, /*reuse_b=*/false);
}

// A fragment: 16x32 bf16. lane<16 -> row = laneLow, elems 0..15 = K k..k+15 ;
// lane>=16 -> K k+16..k+31.
__device__ __forceinline__ bf16x16 load_a_frag(const bf16_t* __restrict__ row,
                                               int kBase, int half) {
  union { bf16x16 v; uint4 q[2]; } f;
  f.q[0] = *(const uint4*)(row + kBase + half * 16);
  f.q[1] = *(const uint4*)(row + kBase + half * 16 + 8);
  return f.v;
}

// Same fragment sourced from an LDS panel row (32 K-values, pitch 32 elems).
__device__ __forceinline__ bf16x16 load_a_frag_lds(const bf16_t* row, int half) {
  union { bf16x16 v; uint4 q[2]; } f;
  f.q[0] = *(const uint4*)(row + half * 16);
  f.q[1] = *(const uint4*)(row + half * 16 + 8);
  return f.v;
}

// B fragment: 32x16 bf16 from a [N][K] (transposed) weight matrix.
// lane holds column n = laneLow, 16 contiguous K values starting at
// kBase + (half ? 16 : 0).
__device__ __forceinline__ bf16x16 load_b_frag(const bf16_t* __restrict__ p) {
  union { bf16x16 v; uint4 q[2]; } f;
  const uint4* q = (const uint4*)p;
  f.q[0] = q[0];
  f.q[1] = q[1];
  return f.v;
}

__device__ __forceinline__ float silu_f(float x) {
  return x / (1.0f + __expf(-x));
}

// ---------------------------------------------------------------------------
// fp32 -> bf16 conversion of x (4 elems / thread)
// ---------------------------------------------------------------------------
__global__ void moe_cvt_x(const float* __restrict__ x, bf16_t* __restrict__ xb, int n) {
  int base = (blockIdx.x * blockDim.x + threadIdx.x) * 4;
  if (base >= n) return;
  float4 v = *(const float4*)(x + base);
  xb[base + 0] = (bf16_t)v.x;
  xb[base + 1] = (bf16_t)v.y;
  xb[base + 2] = (bf16_t)v.z;
  xb[base + 3] = (bf16_t)v.w;
}

// ---------------------------------------------------------------------------
// Transpose + convert: src fp32 [R][C] -> dst bf16 [C][R]. blockDim (32,8),
// grid (C/32, R/32, batch). R,C multiples of 32 here.
// ---------------------------------------------------------------------------
__global__ __launch_bounds__(256) void moe_transpose_cvt(
    const float* __restrict__ src, bf16_t* __restrict__ dst, int R, int C) {
  __shared__ float tile[32][33];
  size_t mat = (size_t)blockIdx.z * R * C;
  const float* s = src + mat;
  bf16_t* d = dst + mat;
  int tx = threadIdx.x, ty = threadIdx.y;
  int col = blockIdx.x * 32 + tx;
#pragma unroll
  for (int j = 0; j < 4; ++j) {
    int row = blockIdx.y * 32 + ty + j * 8;
    tile[ty + j * 8][tx] = s[(size_t)row * C + col];
  }
  __syncthreads();
  int row2 = blockIdx.y * 32 + tx;
#pragma unroll
  for (int j = 0; j < 4; ++j) {
    int col2 = blockIdx.x * 32 + ty + j * 8;
    d[(size_t)col2 * R + row2] = (bf16_t)tile[tx][ty + j * 8];
  }
}

// ---------------------------------------------------------------------------
// Routing: one wave per token. gates = sigmoid(x @ Wg); top-2; normalize;
// fold the final 1/3; build per-expert token-slot lists with atomics.
// ---------------------------------------------------------------------------
__global__ __launch_bounds__(256) void moe_route(
    const float* __restrict__ x, const float* __restrict__ Wg,
    float* __restrict__ tw, int* __restrict__ lists, int* __restrict__ counts) {
  int tok  = (blockIdx.x * blockDim.x + threadIdx.x) >> 5;
  int lane = threadIdx.x & 31;
  if (tok >= T_TOK) return;
  const float* xr = x + (size_t)tok * DMODEL;
  float g[NEXP] = {};
  for (int i = lane; i < DMODEL; i += 32) {
    float xv = xr[i];
    const float* wg = Wg + (size_t)i * NEXP;
#pragma unroll
    for (int e = 0; e < NEXP; ++e) g[e] += xv * wg[e];
  }
#pragma unroll
  for (int e = 0; e < NEXP; ++e)
#pragma unroll
    for (int off = 16; off > 0; off >>= 1) g[e] += __shfl_xor(g[e], off, 32);
  if (lane == 0) {
    float v[NEXP];
#pragma unroll
    for (int e = 0; e < NEXP; ++e) v[e] = 1.0f / (1.0f + __expf(-g[e]));
    int i0 = 0;
#pragma unroll
    for (int e = 1; e < NEXP; ++e) if (v[e] > v[i0]) i0 = e;
    int i1 = (i0 == 0) ? 1 : 0;
#pragma unroll
    for (int e = 0; e < NEXP; ++e) if (e != i0 && v[e] > v[i1]) i1 = e;
    float s  = v[i0] + v[i1];
    float w0 = v[i0] / s * (1.0f / 3.0f);
    float w1 = v[i1] / s * (1.0f / 3.0f);
    int p0 = atomicAdd(&counts[i0], 1);
    lists[i0 * T_TOK + p0] = tok * 2;
    int p1 = atomicAdd(&counts[i1], 1);
    lists[i1 * T_TOK + p1] = tok * 2 + 1;
    tw[tok * 2]     = w0;
    tw[tok * 2 + 1] = w1;
  }
}

// ---------------------------------------------------------------------------
// Shared expert, up: H = silu(X@Ws1) * (X@Ws3)  (bf16 out).
// A-panel (128 rows x 32 K, bf16) staged through LDS by the Tensor Data
// Mover, double-buffered: wave 0 issues TENSOR_LOAD_TO_LDS for panel k+32,
// all waves compute from panel k via ds_load_b128 fragment reads, then
// s_wait_tensorcnt + workgroup barrier.
// ---------------------------------------------------------------------------
__global__ __launch_bounds__(256) void moe_shared_up(
    const bf16_t* __restrict__ X, const bf16_t* __restrict__ W1T,
    const bf16_t* __restrict__ W3T, bf16_t* __restrict__ H) {
  const int K = DMODEL;
  __shared__ __align__(16) bf16_t Apan[2][128][32];
  int lane = threadIdx.x & 31, wave = threadIdx.x >> 5;
  int laneLow = lane & 15, half = lane >> 4;
  int mBlk = blockIdx.x * 128;
  int mW = (wave >> 1) * 32;
  int nW = blockIdx.y * 64 + (wave & 1) * 32;
  f32x8 a1[2][2] = {}, a3[2][2] = {};

  // Issue one TDM descriptor: 2D tile 32(K) x 128(rows) of 2-byte elements,
  // row stride = K elements, into LDS buffer `buf`.  D# packing per ISA
  // 8.3/8.4 (group0: count/lds/global/type, group1: dims/strides).
  // This toolchain exposes the 6-operand builtin:
  //   (uint32x4 g0, int32x8 g1, int32x4 g2, int32x4 g3, int32x8 gx, i32 cpol)
  auto issue_tdm = [&](int kk, int buf) {
    unsigned lds_off = (unsigned)(uintptr_t)(&Apan[buf][0][0]);
    unsigned long long ga =
        (unsigned long long)(uintptr_t)(X + (size_t)mBlk * K + kk);
    u32x4 g0;
    g0[0] = 1u;                                        // count=1, user desc
    g0[1] = lds_off;                                   // lds_addr (bytes)
    g0[2] = (unsigned)(ga & 0xffffffffull);            // global_addr[31:0]
    g0[3] = (unsigned)((ga >> 32) & 0x01ffffffull) |   // global_addr[56:32]
            0x80000000u;                               // type = 2 ("image")
    i32x8 g1;
    g1[0] = 1 << 16;                                   // data_size=1 -> 2B
    g1[1] = (int)(((unsigned)K & 0xffffu) << 16);      // tensor_dim0[15:0]
    g1[2] = (int)((((unsigned)K >> 16) & 0xffffu) |
                  (((unsigned)T_TOK & 0xffffu) << 16));// dim0 hi | dim1 lo
    g1[3] = (int)((((unsigned)T_TOK >> 16) & 0xffffu) |
                  (32u << 16));                        // dim1 hi | tile_dim0=32
    g1[4] = 128;                                       // tile_dim1=128, dim2=0
    g1[5] = K;                                         // dim0_stride[31:0]
    g1[6] = 0;                                         // dim0_stride hi, dim1_stride lo
    g1[7] = 0;                                         // dim1_stride hi
    i32x4 gz  = {0, 0, 0, 0};                          // groups 2/3 unused (2D)
    i32x8 gz8 = {0, 0, 0, 0, 0, 0, 0, 0};             // extra group (unused)
    __builtin_amdgcn_tensor_load_to_lds(g0, g1, gz, gz, gz8, 0);
  };

  if (wave == 0) {
    issue_tdm(0, 0);
    __builtin_amdgcn_s_wait_tensorcnt(0);
  }
  __syncthreads();

  for (int k = 0; k < K; k += 32) {
    int buf = (k >> 5) & 1;
    if (wave == 0 && (k + 32) < K) issue_tdm(k + 32, buf ^ 1);

    bf16x16 fa0 = load_a_frag_lds(&Apan[buf][mW + laneLow][0], half);
    bf16x16 fa1 = load_a_frag_lds(&Apan[buf][mW + 16 + laneLow][0], half);
    int k0 = k + half * 16;
#pragma unroll
    for (int ni = 0; ni < 2; ++ni) {
      size_t n = (size_t)(nW + ni * 16 + laneLow);
      const bf16_t* pb1 = W1T + n * K + k0;
      const bf16_t* pb3 = W3T + n * K + k0;
      __builtin_prefetch(pb1 + 128, 0, 1);
      __builtin_prefetch(pb3 + 128, 0, 1);
      bf16x16 b1 = load_b_frag(pb1);
      bf16x16 b3 = load_b_frag(pb3);
      a1[0][ni] = wmma_bf16(fa0, b1, a1[0][ni]);
      a1[1][ni] = wmma_bf16(fa1, b1, a1[1][ni]);
      a3[0][ni] = wmma_bf16(fa0, b3, a3[0][ni]);
      a3[1][ni] = wmma_bf16(fa1, b3, a3[1][ni]);
    }
    if (wave == 0 && (k + 32) < K) __builtin_amdgcn_s_wait_tensorcnt(0);
    __syncthreads();
  }
#pragma unroll
  for (int mi = 0; mi < 2; ++mi)
#pragma unroll
    for (int ni = 0; ni < 2; ++ni)
#pragma unroll
      for (int r = 0; r < 8; ++r) {
        int m = mBlk + mW + mi * 16 + half * 8 + r;
        int n = nW + ni * 16 + laneLow;
        float h = silu_f(a1[mi][ni][r]) * a3[mi][ni][r];
        H[(size_t)m * FFDIM + n] = (bf16_t)h;
      }
}

// ---------------------------------------------------------------------------
// Shared expert, down: out = (H @ Ws2) / 3.  Wave tile 32x64 (acc[2][4])
// -> 8 WMMA per 6 fragment loads. Block tile 128x128.
// ---------------------------------------------------------------------------
__global__ __launch_bounds__(256) void moe_shared_down(
    const bf16_t* __restrict__ H, const bf16_t* __restrict__ W2T,
    float* __restrict__ out) {
  const int K = FFDIM;
  int lane = threadIdx.x & 31, wave = threadIdx.x >> 5;
  int laneLow = lane & 15, half = lane >> 4;
  int mW = blockIdx.x * 128 + (wave >> 1) * 32;
  int nW = blockIdx.y * 128 + (wave & 1) * 64;
  f32x8 acc[2][4] = {};
  const bf16_t* r0 = H + (size_t)(mW + laneLow) * K;
  const bf16_t* r1 = H + (size_t)(mW + 16 + laneLow) * K;
  for (int k = 0; k < K; k += 32) {
    bf16x16 fa0 = load_a_frag(r0, k, half);
    bf16x16 fa1 = load_a_frag(r1, k, half);
    int k0 = k + half * 16;
#pragma unroll
    for (int ni = 0; ni < 4; ++ni) {
      size_t n = (size_t)(nW + ni * 16 + laneLow);
      const bf16_t* pb = W2T + n * K + k0;
      __builtin_prefetch(pb + 128, 0, 1);
      bf16x16 b = load_b_frag(pb);
      acc[0][ni] = wmma_bf16(fa0, b, acc[0][ni]);
      acc[1][ni] = wmma_bf16(fa1, b, acc[1][ni]);
    }
  }
#pragma unroll
  for (int mi = 0; mi < 2; ++mi)
#pragma unroll
    for (int ni = 0; ni < 4; ++ni)
#pragma unroll
      for (int r = 0; r < 8; ++r) {
        int m = mW + mi * 16 + half * 8 + r;
        int n = nW + ni * 16 + laneLow;
        out[(size_t)m * DMODEL + n] = acc[mi][ni][r] * (1.0f / 3.0f);
      }
}

// ---------------------------------------------------------------------------
// Routed expert, up: per expert e, gathered rows (token lists);
// Hr[tokenslot] = silu(Xg@We1[e]) * (Xg@We3[e]).
// ---------------------------------------------------------------------------
__global__ __launch_bounds__(256) void moe_routed_up(
    const bf16_t* __restrict__ X, const bf16_t* __restrict__ We1T,
    const bf16_t* __restrict__ We3T, const int* __restrict__ lists,
    const int* __restrict__ counts, bf16_t* __restrict__ Hr) {
  const int K = DMODEL;
  int e = blockIdx.z;
  int cnt = counts[e];
  if ((int)(blockIdx.x * 128) >= cnt) return;
  const int* lst = lists + e * T_TOK;
  int lane = threadIdx.x & 31, wave = threadIdx.x >> 5;
  int laneLow = lane & 15, half = lane >> 4;
  int mW = blockIdx.x * 128 + (wave >> 1) * 32;
  int nW = blockIdx.y * 64 + (wave & 1) * 32;
  int m0 = mW + laneLow, m1 = m0 + 16;
  int ts0 = (m0 < cnt) ? lst[m0] : 0;
  int ts1 = (m1 < cnt) ? lst[m1] : 0;
  const bf16_t* r0 = X + (size_t)(ts0 >> 1) * K;
  const bf16_t* r1 = X + (size_t)(ts1 >> 1) * K;
  const bf16_t* W1 = We1T + (size_t)e * FFDIM * DMODEL;
  const bf16_t* W3 = We3T + (size_t)e * FFDIM * DMODEL;
  f32x8 a1[2][2] = {}, a3[2][2] = {};
  for (int k = 0; k < K; k += 32) {
    bf16x16 fa0 = load_a_frag(r0, k, half);
    bf16x16 fa1 = load_a_frag(r1, k, half);
    int k0 = k + half * 16;
#pragma unroll
    for (int ni = 0; ni < 2; ++ni) {
      size_t n = (size_t)(nW + ni * 16 + laneLow);
      const bf16_t* pb1 = W1 + n * K + k0;
      const bf16_t* pb3 = W3 + n * K + k0;
      __builtin_prefetch(pb1 + 128, 0, 1);
      __builtin_prefetch(pb3 + 128, 0, 1);
      bf16x16 b1 = load_b_frag(pb1);
      bf16x16 b3 = load_b_frag(pb3);
      a1[0][ni] = wmma_bf16(fa0, b1, a1[0][ni]);
      a1[1][ni] = wmma_bf16(fa1, b1, a1[1][ni]);
      a3[0][ni] = wmma_bf16(fa0, b3, a3[0][ni]);
      a3[1][ni] = wmma_bf16(fa1, b3, a3[1][ni]);
    }
  }
#pragma unroll
  for (int mi = 0; mi < 2; ++mi)
#pragma unroll
    for (int ni = 0; ni < 2; ++ni)
#pragma unroll
      for (int r = 0; r < 8; ++r) {
        int m = mW + mi * 16 + half * 8 + r;
        if (m < cnt) {
          int ts = lst[m];
          int n = nW + ni * 16 + laneLow;
          float h = silu_f(a1[mi][ni][r]) * a3[mi][ni][r];
          Hr[(size_t)ts * FFDIM + n] = (bf16_t)h;
        }
      }
}

// ---------------------------------------------------------------------------
// Routed expert, down: out[token] += w * (Hr[tokenslot] @ We2[e]).
// Wave tile 32x64, block tile 128x128.
// ---------------------------------------------------------------------------
__global__ __launch_bounds__(256) void moe_routed_down(
    const bf16_t* __restrict__ Hr, const bf16_t* __restrict__ We2T,
    const int* __restrict__ lists, const int* __restrict__ counts,
    const float* __restrict__ tw, float* __restrict__ out) {
  const int K = FFDIM;
  int e = blockIdx.z;
  int cnt = counts[e];
  if ((int)(blockIdx.x * 128) >= cnt) return;
  const int* lst = lists + e * T_TOK;
  int lane = threadIdx.x & 31, wave = threadIdx.x >> 5;
  int laneLow = lane & 15, half = lane >> 4;
  int mW = blockIdx.x * 128 + (wave >> 1) * 32;
  int nW = blockIdx.y * 128 + (wave & 1) * 64;
  int m0 = mW + laneLow, m1 = m0 + 16;
  int ts0 = (m0 < cnt) ? lst[m0] : 0;
  int ts1 = (m1 < cnt) ? lst[m1] : 0;
  const bf16_t* r0 = Hr + (size_t)ts0 * K;
  const bf16_t* r1 = Hr + (size_t)ts1 * K;
  const bf16_t* W2 = We2T + (size_t)e * DMODEL * FFDIM;
  f32x8 acc[2][4] = {};
  for (int k = 0; k < K; k += 32) {
    bf16x16 fa0 = load_a_frag(r0, k, half);
    bf16x16 fa1 = load_a_frag(r1, k, half);
    int k0 = k + half * 16;
#pragma unroll
    for (int ni = 0; ni < 4; ++ni) {
      size_t n = (size_t)(nW + ni * 16 + laneLow);
      const bf16_t* pb = W2 + n * K + k0;
      __builtin_prefetch(pb + 128, 0, 1);
      bf16x16 b = load_b_frag(pb);
      acc[0][ni] = wmma_bf16(fa0, b, acc[0][ni]);
      acc[1][ni] = wmma_bf16(fa1, b, acc[1][ni]);
    }
  }
#pragma unroll
  for (int mi = 0; mi < 2; ++mi)
#pragma unroll
    for (int r = 0; r < 8; ++r) {
      int m = mW + mi * 16 + half * 8 + r;
      if (m >= cnt) continue;
      int ts = lst[m];
      int token = ts >> 1;
      float w = tw[ts];
#pragma unroll
      for (int ni = 0; ni < 4; ++ni) {
        int n = nW + ni * 16 + laneLow;
        atomicAdd(&out[(size_t)token * DMODEL + n], acc[mi][ni][r] * w);
      }
    }
}

// ---------------------------------------------------------------------------
// Host-side orchestration
// ---------------------------------------------------------------------------
extern "C" void kernel_launch(void* const* d_in, const int* in_sizes, int n_in,
                              void* d_out, int out_size, void* d_ws, size_t ws_size,
                              hipStream_t stream) {
  (void)in_sizes; (void)n_in; (void)out_size; (void)ws_size;
  const float* x   = (const float*)d_in[0];
  const float* Wg  = (const float*)d_in[1];
  const float* Ws1 = (const float*)d_in[2];
  const float* Ws3 = (const float*)d_in[3];
  const float* Ws2 = (const float*)d_in[4];
  const float* We1 = (const float*)d_in[5];
  const float* We3 = (const float*)d_in[6];
  const float* We2 = (const float*)d_in[7];
  float* out = (float*)d_out;

  char* ws = (char*)d_ws;
  size_t off = 0;
  auto take = [&](size_t bytes) -> char* {
    char* p = ws + off;
    off += (bytes + 255) & ~(size_t)255;
    return p;
  };
  bf16_t* xb    = (bf16_t*)take((size_t)T_TOK * DMODEL * 2);
  bf16_t* Ws1T  = (bf16_t*)take((size_t)FFDIM * DMODEL * 2);
  bf16_t* Ws3T  = (bf16_t*)take((size_t)FFDIM * DMODEL * 2);
  bf16_t* Ws2T  = (bf16_t*)take((size_t)DMODEL * FFDIM * 2);
  bf16_t* We1T  = (bf16_t*)take((size_t)NEXP * FFDIM * DMODEL * 2);
  bf16_t* We3T  = (bf16_t*)take((size_t)NEXP * FFDIM * DMODEL * 2);
  bf16_t* We2T  = (bf16_t*)take((size_t)NEXP * DMODEL * FFDIM * 2);
  bf16_t* Hs    = (bf16_t*)take((size_t)T_TOK * FFDIM * 2);
  bf16_t* Hr    = (bf16_t*)take((size_t)T_TOK * 2 * FFDIM * 2);
  int*    counts= (int*)   take(NEXP * sizeof(int));
  int*    lists = (int*)   take((size_t)NEXP * T_TOK * sizeof(int));
  float*  tw    = (float*) take((size_t)T_TOK * 2 * sizeof(float));

  (void)hipMemsetAsync(counts, 0, NEXP * sizeof(int), stream);

  // x -> bf16
  {
    int n = T_TOK * DMODEL;
    moe_cvt_x<<<dim3(n / 4 / 256), dim3(256), 0, stream>>>(x, xb, n);
  }
  // weights -> transposed bf16 [N][K]
  dim3 tb(32, 8);
  moe_transpose_cvt<<<dim3(FFDIM / 32, DMODEL / 32, 1),    tb, 0, stream>>>(Ws1, Ws1T, DMODEL, FFDIM);
  moe_transpose_cvt<<<dim3(FFDIM / 32, DMODEL / 32, 1),    tb, 0, stream>>>(Ws3, Ws3T, DMODEL, FFDIM);
  moe_transpose_cvt<<<dim3(DMODEL / 32, FFDIM / 32, 1),    tb, 0, stream>>>(Ws2, Ws2T, FFDIM, DMODEL);
  moe_transpose_cvt<<<dim3(FFDIM / 32, DMODEL / 32, NEXP), tb, 0, stream>>>(We1, We1T, DMODEL, FFDIM);
  moe_transpose_cvt<<<dim3(FFDIM / 32, DMODEL / 32, NEXP), tb, 0, stream>>>(We3, We3T, DMODEL, FFDIM);
  moe_transpose_cvt<<<dim3(DMODEL / 32, FFDIM / 32, NEXP), tb, 0, stream>>>(We2, We2T, FFDIM, DMODEL);

  // routing (after counts memset)
  moe_route<<<dim3((T_TOK * 32) / 256), dim3(256), 0, stream>>>(x, Wg, tw, lists, counts);

  // shared expert
  moe_shared_up  <<<dim3(T_TOK / 128, FFDIM / 64),   dim3(256), 0, stream>>>(xb, Ws1T, Ws3T, Hs);
  moe_shared_down<<<dim3(T_TOK / 128, DMODEL / 128), dim3(256), 0, stream>>>(Hs, Ws2T, out);

  // routed experts (sparse, only selected tokens per expert)
  moe_routed_up  <<<dim3(T_TOK / 128, FFDIM / 64,   NEXP), dim3(256), 0, stream>>>(
      xb, We1T, We3T, lists, counts, Hr);
  moe_routed_down<<<dim3(T_TOK / 128, DMODEL / 128, NEXP), dim3(256), 0, stream>>>(
      Hr, We2T, lists, counts, tw, out);
}
